// OutProj_80290118632066
// MI455X (gfx1250) — matrix operand order repaired and checked by reading, not compiled, not run on hardware
//
#include <hip/hip_runtime.h>

// Quantized int4 GEMM + fused epilogue for MI455X (gfx1250, wave32, WMMA).
// Strategy: prepass dequant/convert to bf16 in workspace (L2-resident),
// then a pure bf16 WMMA GEMM with async global->LDS double buffering.
// Main GEMM: 128 threads (4 waves), 64x64 wave tile -> 1.0 ds_load per WMMA.

#define BDIM  256            // fallback kernel block size
#define BDIM2 128            // main GEMM block size (4 waves)
#define BM 128
#define BN 128
#define BK 64
#define LDA 72               // padded LDS row stride (bf16 elems); 144B, 16B aligned
#define M_TOT 8192
#define N_TOT 4096
#define K_TOT 4096
#define GQ 128

typedef __attribute__((ext_vector_type(4)))  int    v4i;
typedef __attribute__((ext_vector_type(8)))  __bf16 v8bf;
typedef __attribute__((ext_vector_type(16))) __bf16 v16bf;
typedef __attribute__((ext_vector_type(8)))  float  v8f;

#if __has_builtin(__builtin_amdgcn_global_load_async_to_lds_b128)
#define HAVE_ASYNC_LDS 1
#else
#define HAVE_ASYNC_LDS 0
#endif

// round-to-nearest-even f32 -> bf16, packed pairwise into a u32
__device__ __forceinline__ unsigned int pack2bf(float a, float b) {
    unsigned int ua = __float_as_uint(a);
    unsigned int ub = __float_as_uint(b);
    ua = (ua + 0x7FFFu + ((ua >> 16) & 1u)) >> 16;
    ub = (ub + 0x7FFFu + ((ub >> 16) & 1u)) >> 16;
    return (ua & 0xFFFFu) | (ub << 16);
}

// 16-byte global -> LDS copy; async DMA path on gfx1250 if the builtin exists.
__device__ __forceinline__ void copy16_g2l(const unsigned short* g, unsigned short* l) {
#if HAVE_ASYNC_LDS
    unsigned short* gnc = (unsigned short*)g;  // builtin param is non-const
    __builtin_amdgcn_global_load_async_to_lds_b128(
        (__attribute__((address_space(1))) v4i*)gnc,
        (__attribute__((address_space(3))) v4i*)l, 0, 0);
#else
    *(uint4*)l = *(const uint4*)g;
#endif
}

__device__ __forceinline__ void wait_async0() {
#if HAVE_ASYNC_LDS
#if __has_builtin(__builtin_amdgcn_s_wait_asynccnt)
    __builtin_amdgcn_s_wait_asynccnt(0);
#else
    asm volatile("s_wait_asynccnt 0" ::: "memory");
#endif
#endif
}

// ---------------- prepass 1: x fp32 -> bf16 ----------------
__global__ __launch_bounds__(256) void cvt_x_bf16(const float* __restrict__ x,
                                                  unsigned short* __restrict__ xb) {
    const size_t i = ((size_t)blockIdx.x * 256 + threadIdx.x) * 4;
    const float4 f = *(const float4*)(x + i);
    *(uint2*)(xb + i) = make_uint2(pack2bf(f.x, f.y), pack2bf(f.z, f.w));
}

// ---------------- prepass 2: packed int4 -> bf16 weights ----------------
__global__ __launch_bounds__(256) void dequant_w_bf16(const int* __restrict__ w4,
                                                      const float* __restrict__ wscale,
                                                      unsigned short* __restrict__ wb) {
    const size_t i = (size_t)blockIdx.x * 256 + threadIdx.x;  // word index, N*K/8 total
    const int w    = w4[i];
    const int row  = (int)(i >> 9);          // K/8 = 512 words per row
    const int wj   = (int)(i & 511);         // word within row; k = wj*8
    const float s  = wscale[(size_t)row * (K_TOT / GQ) + (wj >> 4)];  // 16 words/group
    unsigned int pk[4];
    #pragma unroll
    for (int q = 0; q < 4; ++q) {
        const int n0 = (w << (28 - 8 * q)) >> 28;   // nibble 2q
        const int n1 = (w << (24 - 8 * q)) >> 28;   // nibble 2q+1
        pk[q] = pack2bf((float)n0 * s, (float)n1 * s);
    }
    *(uint4*)(wb + i * 8) = make_uint4(pk[0], pk[1], pk[2], pk[3]);
}

// Stage one 128x64 bf16 tile (row stride K_TOT in global) into padded LDS.
// 1024 x 16B chunks; 8 async instructions per thread with 128 threads.
__device__ __forceinline__ void stage_tile128(const unsigned short* __restrict__ gbase,
                                              unsigned short* lbase, int tid) {
    #pragma unroll
    for (int inst = 0; inst < 8; ++inst) {
        const int c   = inst * BDIM2 + tid;
        const int row = c >> 3;
        const int kc  = (c & 7) * 8;
        copy16_g2l(gbase + (size_t)row * K_TOT + kc, lbase + row * LDA + kc);
    }
}

// ---------------- main bf16 WMMA GEMM, async double-buffered ----------------
// 4 waves, each owns a 64x64 output tile: 4x4 WMMA accumulators.
__global__ __launch_bounds__(BDIM2) void wmma_gemm_bf16(
    const unsigned short* __restrict__ xb,      // [M, K] bf16
    const unsigned short* __restrict__ wb,      // [N, K] bf16
    const float* __restrict__ x_scale,          // [M]
    const float* __restrict__ residual,         // [M, N]
    const float* __restrict__ bias,             // [N]
    float*       __restrict__ out)              // [M, N]
{
    __shared__ __align__(16) unsigned short As[2][BM * LDA];
    __shared__ __align__(16) unsigned short Bs[2][BN * LDA];

    const int tid    = threadIdx.x;
    const int lane   = tid & 31;
    const int wave   = tid >> 5;        // 4 waves
    const int wave_m = wave & 1;        // 2 waves along M (64 rows each)
    const int wave_n = wave >> 1;       // 2 waves along N (64 cols each)

    const int tileN = blockIdx.x * BN;
    const int tileM = blockIdx.y * BM;

    const int lrow = lane & 15;
    const int lhi  = lane >> 4;

    v8f acc[4][4] = {};

    const unsigned short* agbase = xb + (size_t)tileM * K_TOT;
    const unsigned short* bgbase = wb + (size_t)tileN * K_TOT;

    // prologue: stage first tiles into buffer 0
    stage_tile128(agbase, As[0], tid);
    stage_tile128(bgbase, Bs[0], tid);

    int buf = 0;
    for (int k0 = 0; k0 < K_TOT; k0 += BK) {
        wait_async0();       // my async writes to buf are done
        __syncthreads();     // everyone's are done

        if (k0 + BK < K_TOT) {   // stream next tiles into other buffer
            stage_tile128(agbase + (k0 + BK), As[buf ^ 1], tid);
            stage_tile128(bgbase + (k0 + BK), Bs[buf ^ 1], tid);
        }

        #pragma unroll
        for (int kk = 0; kk < BK; kk += 32) {
            v16bf afrag[4];
            v16bf bfrag[4];
            #pragma unroll
            for (int mt = 0; mt < 4; ++mt) {
                // A 16x32 bf16 layout: lane = M-row; runs [kk+lhi*8,+8) and +16
                const unsigned short* ap =
                    &As[buf][(wave_m * 64 + mt * 16 + lrow) * LDA + kk + lhi * 8];
                union { v16bf v; v8bf h[2]; } ua;
                ua.h[0] = *(const v8bf*)(ap);
                ua.h[1] = *(const v8bf*)(ap + 16);
                afrag[mt] = ua.v;
            }
            #pragma unroll
            for (int nt = 0; nt < 4; ++nt) {
                // B 32x16 bf16 layout: lane = N-col; runs [kk+lhi*8,+8) and +16
                const unsigned short* bp =
                    &Bs[buf][(wave_n * 64 + nt * 16 + lrow) * LDA + kk + lhi * 8];
                union { v16bf v; v8bf h[2]; } ub;
                ub.h[0] = *(const v8bf*)(bp);
                ub.h[1] = *(const v8bf*)(bp + 16);
                bfrag[nt] = ub.v;
            }
            #pragma unroll
            for (int mt = 0; mt < 4; ++mt)
                #pragma unroll
                for (int nt = 0; nt < 4; ++nt)
                    acc[mt][nt] = __builtin_amdgcn_wmma_f32_16x16x32_bf16(
                        false, afrag[mt], false, bfrag[nt],
                        (short)0, acc[mt][nt], false, false);
        }
        __syncthreads();
        buf ^= 1;
    }

    // fused epilogue: *x_scale + residual + bias
    #pragma unroll
    for (int mt = 0; mt < 4; ++mt) {
        #pragma unroll
        for (int nt = 0; nt < 4; ++nt) {
            #pragma unroll
            for (int r = 0; r < 8; ++r) {
                const int row = tileM + wave_m * 64 + mt * 16 + lhi * 8 + r;
                const int col = tileN + wave_n * 64 + nt * 16 + lrow;
                const size_t idx = (size_t)row * N_TOT + col;
                out[idx] = acc[mt][nt][r] * x_scale[row] + residual[idx] + bias[col];
            }
        }
    }
}

// ---------------- fallback: fully fused kernel (no workspace needed) ----------------
__global__ __launch_bounds__(BDIM) void wmma_gemm_i4_fused(
    const float* __restrict__ x, const float* __restrict__ x_scale,
    const int* __restrict__ w4, const float* __restrict__ wscale,
    const float* __restrict__ residual, const float* __restrict__ bias,
    float* __restrict__ out)
{
    __shared__ __align__(16) unsigned short As[BM * LDA];
    __shared__ __align__(16) unsigned short Bs[BN * LDA];

    const int tid = threadIdx.x, lane = tid & 31, wave = tid >> 5;
    const int wave_m = wave & 1, wave_n = wave >> 1;
    const int tileN = blockIdx.x * BN, tileM = blockIdx.y * BM;
    const int a_row = tid >> 4, a_col = (tid & 15) << 2;
    const int b_row = tid >> 1, b_half = tid & 1;
    const int KW = K_TOT / 8;
    const int lrow = lane & 15, lhi = lane >> 4;
    v8f acc[4][2] = {};

    for (int k0 = 0; k0 < K_TOT; k0 += BK) {
        #pragma unroll
        for (int p = 0; p < 8; ++p) {
            const int r = p * 16 + a_row;
            const float4 f = *(const float4*)(x + (size_t)(tileM + r) * K_TOT + k0 + a_col);
            *(uint2*)(&As[r * LDA + a_col]) = make_uint2(pack2bf(f.x, f.y), pack2bf(f.z, f.w));
        }
        {
            const int4 w = *(const int4*)(w4 + (size_t)(tileN + b_row) * KW + (k0 >> 3) + b_half * 4);
            const float s = wscale[(size_t)(tileN + b_row) * (K_TOT / GQ) + (k0 >> 7)];
            const int wv[4] = { w.x, w.y, w.z, w.w };
            #pragma unroll
            for (int j = 0; j < 4; ++j) {
                unsigned int pk[4];
                #pragma unroll
                for (int q = 0; q < 4; ++q) {
                    const int n0 = (wv[j] << (28 - 8 * q)) >> 28;
                    const int n1 = (wv[j] << (24 - 8 * q)) >> 28;
                    pk[q] = pack2bf((float)n0 * s, (float)n1 * s);
                }
                *(uint4*)(&Bs[b_row * LDA + b_half * 32 + j * 8]) = make_uint4(pk[0], pk[1], pk[2], pk[3]);
            }
        }
        __syncthreads();
        #pragma unroll
        for (int kk = 0; kk < BK; kk += 32) {
            v16bf afrag[4]; v16bf bfrag[2];
            #pragma unroll
            for (int mt = 0; mt < 4; ++mt) {
                const unsigned short* ap = &As[(wave_m * 64 + mt * 16 + lrow) * LDA + kk + lhi * 8];
                union { v16bf v; v8bf h[2]; } ua;
                ua.h[0] = *(const v8bf*)(ap); ua.h[1] = *(const v8bf*)(ap + 16);
                afrag[mt] = ua.v;
            }
            #pragma unroll
            for (int nt = 0; nt < 2; ++nt) {
                const unsigned short* bp = &Bs[(wave_n * 32 + nt * 16 + lrow) * LDA + kk + lhi * 8];
                union { v16bf v; v8bf h[2]; } ub;
                ub.h[0] = *(const v8bf*)(bp); ub.h[1] = *(const v8bf*)(bp + 16);
                bfrag[nt] = ub.v;
            }
            #pragma unroll
            for (int mt = 0; mt < 4; ++mt)
                #pragma unroll
                for (int nt = 0; nt < 2; ++nt)
                    acc[mt][nt] = __builtin_amdgcn_wmma_f32_16x16x32_bf16(
                        false, afrag[mt], false, bfrag[nt], (short)0, acc[mt][nt], false, false);
        }
        __syncthreads();
    }
    #pragma unroll
    for (int mt = 0; mt < 4; ++mt)
        #pragma unroll
        for (int nt = 0; nt < 2; ++nt)
            #pragma unroll
            for (int r = 0; r < 8; ++r) {
                const int row = tileM + wave_m * 64 + mt * 16 + lhi * 8 + r;
                const int col = tileN + wave_n * 32 + nt * 16 + lrow;
                const size_t idx = (size_t)row * N_TOT + col;
                out[idx] = acc[mt][nt][r] * x_scale[row] + residual[idx] + bias[col];
            }
}

extern "C" void kernel_launch(void* const* d_in, const int* in_sizes, int n_in,
                              void* d_out, int out_size, void* d_ws, size_t ws_size,
                              hipStream_t stream) {
    (void)in_sizes; (void)n_in; (void)out_size;
    const float* x        = (const float*)d_in[0];
    const float* x_scale  = (const float*)d_in[1];
    const int*   w4       = (const int*)d_in[2];
    const float* wscale   = (const float*)d_in[3];
    const float* residual = (const float*)d_in[4];
    const float* bias     = (const float*)d_in[5];
    float* out = (float*)d_out;

    const size_t xb_bytes = (size_t)M_TOT * K_TOT * 2;
    const size_t wb_bytes = (size_t)N_TOT * K_TOT * 2;
    dim3 grid(N_TOT / BN, M_TOT / BM);   // (32, 64)

    if (ws_size >= xb_bytes + wb_bytes) {
        unsigned short* xb = (unsigned short*)d_ws;
        unsigned short* wb = xb + (size_t)M_TOT * K_TOT;
        cvt_x_bf16<<<(M_TOT * (size_t)K_TOT / 4) / 256, 256, 0, stream>>>(x, xb);
        dequant_w_bf16<<<((size_t)N_TOT * K_TOT / 8) / 256, 256, 0, stream>>>(w4, wscale, wb);
        wmma_gemm_bf16<<<grid, BDIM2, 0, stream>>>(xb, wb, x_scale, residual, bias, out);
    } else {
        wmma_gemm_i4_fused<<<grid, BDIM, 0, stream>>>(x, x_scale, w4, wscale,
                                                      residual, bias, out);
    }
}
